// RTDetrMSDeformableAttention_14499809591612
// MI455X (gfx1250) — matrix operand order repaired and checked by reading, not compiled, not run on hardware
//
#include <hip/hip_runtime.h>
#include <hip/hip_bf16.h>

// ---------- constants from the reference ----------
#define BB 32
#define QQ 300
#define DD 256
#define HH 8
#define LL 3
#define PP 4
#define HD 32
#define LEN_V 8400           // 80*80 + 40*40 + 20*20
#define NOFF (HH*LL*PP*2)    // 192
#define NATT (HH*LL*PP)      // 96

typedef __bf16 v16bf __attribute__((ext_vector_type(16)));
typedef float  v8f   __attribute__((ext_vector_type(8)));

// =====================================================================
// Pack a 256x256 fp32 weight matrix into bf16 WMMA B-fragment order.
// Fragment (kt 0..7, nt 0..15): lane l, element e ->
//   K = kt*32 + (l>>4)*8 + (e&7) + (e>>3)*16 ,  N = nt*16 + (l&15)
// Packed layout: Bp[((kt*16+nt)*32 + lane)*16 + e]  (each lane's 16 bf16
// contiguous -> one 32B v16bf load in the GEMM hot loop).
// =====================================================================
__global__ __launch_bounds__(256)
void pack_b_kernel(const float* __restrict__ Bw, __bf16* __restrict__ Bp) {
  const int idx  = blockIdx.x * 256 + threadIdx.x;   // 0..65535
  const int e    = idx & 15;
  const int lane = (idx >> 4) & 31;
  const int frag = idx >> 9;                          // kt*16 + nt
  const int nt   = frag & 15;
  const int kt   = frag >> 4;
  const int K = kt * 32 + (lane >> 4) * 8 + (e & 7) + (e >> 3) * 16;
  const int N = nt * 16 + (lane & 15);
  Bp[idx] = (__bf16)Bw[(size_t)K * DD + N];
}

// ---- load one 16x32 bf16 A fragment (rows row0..row0+15, K kbase-block) ----
__device__ __forceinline__ v16bf load_afrag(const float* __restrict__ arow, int kbase) {
  const float4* lo = (const float4*)(arow + kbase);
  const float4* hi = (const float4*)(arow + kbase + 16);
  const float4 l0 = lo[0], l1 = lo[1], h0 = hi[0], h1 = hi[1];
  v16bf a;
  a[0]  = (__bf16)l0.x; a[1]  = (__bf16)l0.y; a[2]  = (__bf16)l0.z; a[3]  = (__bf16)l0.w;
  a[4]  = (__bf16)l1.x; a[5]  = (__bf16)l1.y; a[6]  = (__bf16)l1.z; a[7]  = (__bf16)l1.w;
  a[8]  = (__bf16)h0.x; a[9]  = (__bf16)h0.y; a[10] = (__bf16)h0.z; a[11] = (__bf16)h0.w;
  a[12] = (__bf16)h1.x; a[13] = (__bf16)h1.y; a[14] = (__bf16)h1.z; a[15] = (__bf16)h1.w;
  return a;
}

// =====================================================================
// WMMA GEMM:  C[M x 256] = A[M x 256] * B[256 x 256] + bias[256]
// B is pre-packed bf16 (fragment order). Block = 128 threads = 4 waves.
// Block computes a 32-row strip; wave w owns columns [w*64, w*64+64).
// Each wave: 2 A-fragments (rows +0, +16) share every B fragment -> 8
// v_wmma per 32-wide K step.
// =====================================================================
__global__ __launch_bounds__(128)
void wmma_gemm256_kernel(const float* __restrict__ A,
                         const __bf16* __restrict__ Bp,
                         const float* __restrict__ bias,
                         float* __restrict__ C, int M) {
  const int lane  = threadIdx.x & 31;
  const int wave  = threadIdx.x >> 5;          // 0..3
  const int row0  = blockIdx.x * 32;
  const int n0    = wave * 64;
  const int rlane = lane & 15;
  const int half  = lane >> 4;                 // 0 or 1
  const int nt0   = wave * 4;                  // first of this wave's 4 N-tiles

  if (row0 >= M) return;

  v8f accA0 = {}, accA1 = {}, accA2 = {}, accA3 = {};
  v8f accB0 = {}, accB1 = {}, accB2 = {}, accB3 = {};

  const float* arow0 = A + (size_t)(row0 + rlane) * DD;
  const float* arow1 = arow0 + (size_t)16 * DD;
  const v16bf* bfrag = (const v16bf*)Bp + lane;   // + (kt*16+nt)*32

  for (int kt = 0; kt < 8; ++kt) {
    const int kbase = kt * 32 + half * 8;
    const v16bf a0 = load_afrag(arow0, kbase);
    const v16bf a1 = load_afrag(arow1, kbase);
    const v16bf* bp = bfrag + (size_t)(kt * 16 + nt0) * 32;
    {
      const v16bf b = bp[0 * 32];
      accA0 = __builtin_amdgcn_wmma_f32_16x16x32_bf16(false, a0, false, b, (short)0, accA0, false, false);
      accB0 = __builtin_amdgcn_wmma_f32_16x16x32_bf16(false, a1, false, b, (short)0, accB0, false, false);
    }
    {
      const v16bf b = bp[1 * 32];
      accA1 = __builtin_amdgcn_wmma_f32_16x16x32_bf16(false, a0, false, b, (short)0, accA1, false, false);
      accB1 = __builtin_amdgcn_wmma_f32_16x16x32_bf16(false, a1, false, b, (short)0, accB1, false, false);
    }
    {
      const v16bf b = bp[2 * 32];
      accA2 = __builtin_amdgcn_wmma_f32_16x16x32_bf16(false, a0, false, b, (short)0, accA2, false, false);
      accB2 = __builtin_amdgcn_wmma_f32_16x16x32_bf16(false, a1, false, b, (short)0, accB2, false, false);
    }
    {
      const v16bf b = bp[3 * 32];
      accA3 = __builtin_amdgcn_wmma_f32_16x16x32_bf16(false, a0, false, b, (short)0, accA3, false, false);
      accB3 = __builtin_amdgcn_wmma_f32_16x16x32_bf16(false, a1, false, b, (short)0, accB3, false, false);
    }
  }

  // ---- store D: VGPR r -> row = base + r + half*8, col = n0 + nt*16 + rlane
  #pragma unroll
  for (int nt = 0; nt < 4; ++nt) {
    const int n = n0 + nt * 16 + rlane;
    const float bv = bias[n];
    const v8f aA = (nt == 0) ? accA0 : (nt == 1) ? accA1 : (nt == 2) ? accA2 : accA3;
    const v8f aB = (nt == 0) ? accB0 : (nt == 1) ? accB1 : (nt == 2) ? accB2 : accB3;
    #pragma unroll
    for (int r = 0; r < 8; ++r) {
      const int m = row0 + r + half * 8;
      C[(size_t)m * DD + n]        = aA[r] + bv;
      C[(size_t)(m + 16) * DD + n] = aB[r] + bv;
    }
  }
}

// =====================================================================
// Query projections: off = q@w_off + b_off ; aw = softmax(q@w_attn+b_attn)
// One block per (b,q) row. 288 threads = 192 off cols + 96 attn cols.
// =====================================================================
__global__ __launch_bounds__(288)
void qproj_kernel(const float* __restrict__ query,
                  const float* __restrict__ w_off, const float* __restrict__ b_off,
                  const float* __restrict__ w_attn, const float* __restrict__ b_attn,
                  float* __restrict__ off_out, float* __restrict__ aw_out) {
  __shared__ float qrow[DD];
  __shared__ float logits[NATT];
  const int row = blockIdx.x;                 // b*Q + q
  const int t = threadIdx.x;
  if (t < DD) qrow[t] = query[(size_t)row * DD + t];
  __syncthreads();

  if (t < NOFF) {
    float s = b_off[t];
    #pragma unroll 8
    for (int k = 0; k < DD; ++k) s += qrow[k] * w_off[(size_t)k * NOFF + t];
    off_out[(size_t)row * NOFF + t] = s;
  } else {
    const int c = t - NOFF;                   // 0..95
    float s = b_attn[c];
    #pragma unroll 8
    for (int k = 0; k < DD; ++k) s += qrow[k] * w_attn[(size_t)k * NATT + c];
    logits[c] = s;
  }
  __syncthreads();

  if (t < HH) {                               // softmax over 12 per head
    float mx = -1e30f;
    #pragma unroll
    for (int i = 0; i < LL * PP; ++i) mx = fmaxf(mx, logits[t * 12 + i]);
    float e[LL * PP];
    float se = 0.f;
    #pragma unroll
    for (int i = 0; i < LL * PP; ++i) { e[i] = __expf(logits[t * 12 + i] - mx); se += e[i]; }
    const float inv = 1.f / se;
    #pragma unroll
    for (int i = 0; i < LL * PP; ++i) aw_out[(size_t)row * NATT + t * 12 + i] = e[i] * inv;
  }
}

// =====================================================================
// Deformable sampling: one block per (b,q); 256 threads, tid = h*32 + d.
// Each wave covers exactly one head -> ref/off/aw are wave-uniform and the
// 4-corner gathers from v are 128B coalesced across the wave.
// =====================================================================
__global__ __launch_bounds__(256)
void msda_sample_kernel(const float* __restrict__ refp,
                        const float* __restrict__ off_ws,
                        const float* __restrict__ aw_ws,
                        const float* __restrict__ v_ws,
                        float* __restrict__ out) {
  const int row = blockIdx.x;                 // b*Q + q
  const int b = row / QQ;
  const int t = threadIdx.x;
  const int h = t >> 5;
  const int d = t & 31;

  const int lvlDim[3]   = {80, 40, 20};
  const int lvlStart[3] = {0, 6400, 8000};

  const float* ref = refp   + (size_t)row * (LL * 2);
  const float* off = off_ws + (size_t)row * NOFF + h * (LL * PP * 2);
  const float* aw  = aw_ws  + (size_t)row * NATT + h * (LL * PP);

  float acc = 0.f;
  #pragma unroll
  for (int l = 0; l < LL; ++l) {
    const float rx = ref[l * 2 + 0];
    const float ry = ref[l * 2 + 1];
    const int   Wi = lvlDim[l];
    const float Wf = (float)Wi;
    const float* vbase = v_ws + ((size_t)b * LEN_V + lvlStart[l]) * DD + h * HD + d;
    #pragma unroll
    for (int p = 0; p < PP; ++p) {
      const float ox = off[(l * PP + p) * 2 + 0];
      const float oy = off[(l * PP + p) * 2 + 1];
      // x = loc_x * W - 0.5  (grid math collapses to this; square levels H==W)
      const float x = (rx + ox / Wf) * Wf - 0.5f;
      const float y = (ry + oy / Wf) * Wf - 0.5f;
      const float x0f = floorf(x), y0f = floorf(y);
      const float wx1 = x - x0f,  wy1 = y - y0f;
      const float wx0 = 1.f - wx1, wy0 = 1.f - wy1;
      const int x0 = (int)x0f, y0 = (int)y0f;
      const int x1 = x0 + 1,   y1 = y0 + 1;
      float s = 0.f;
      if (y0 >= 0 && y0 < Wi) {
        if (x0 >= 0 && x0 < Wi) s += wy0 * wx0 * vbase[(size_t)(y0 * Wi + x0) * DD];
        if (x1 >= 0 && x1 < Wi) s += wy0 * wx1 * vbase[(size_t)(y0 * Wi + x1) * DD];
      }
      if (y1 >= 0 && y1 < Wi) {
        if (x0 >= 0 && x0 < Wi) s += wy1 * wx0 * vbase[(size_t)(y1 * Wi + x0) * DD];
        if (x1 >= 0 && x1 < Wi) s += wy1 * wx1 * vbase[(size_t)(y1 * Wi + x1) * DD];
      }
      acc += aw[l * PP + p] * s;
    }
  }
  out[(size_t)row * DD + t] = acc;            // col = h*HD + d, matches reference
}

// =====================================================================
// Launch
// =====================================================================
extern "C" void kernel_launch(void* const* d_in, const int* in_sizes, int n_in,
                              void* d_out, int out_size, void* d_ws, size_t ws_size,
                              hipStream_t stream) {
  const float* query  = (const float*)d_in[0];
  const float* refp   = (const float*)d_in[1];
  const float* value  = (const float*)d_in[2];
  // d_in[3] = value_spatial_shapes (int64) - static, hardcoded
  const float* w_off  = (const float*)d_in[4];
  const float* b_off  = (const float*)d_in[5];
  const float* w_attn = (const float*)d_in[6];
  const float* b_attn = (const float*)d_in[7];
  const float* w_val  = (const float*)d_in[8];
  const float* b_val  = (const float*)d_in[9];
  const float* w_out  = (const float*)d_in[10];
  const float* b_out  = (const float*)d_in[11];
  float* out = (float*)d_out;

  // workspace layout (floats, all 32B-aligned offsets)
  float* ws = (float*)d_ws;
  const size_t n_v   = (size_t)BB * LEN_V * DD;   // 68,812,800
  const size_t n_off = (size_t)BB * QQ * NOFF;    //  1,843,200
  const size_t n_aw  = (size_t)BB * QQ * NATT;    //    921,600
  const size_t n_att = (size_t)BB * QQ * DD;      //  2,457,600
  float* v_ws   = ws;
  float* off_ws = v_ws + n_v;
  float* aw_ws  = off_ws + n_off;
  float* att_ws = aw_ws + n_aw;
  __bf16* packB1 = (__bf16*)(att_ws + n_att);     // 65536 bf16 = 128KB
  __bf16* packB2 = packB1 + (size_t)DD * DD;

  const int M1 = BB * LEN_V;                      // 268,800 rows
  const int M2 = BB * QQ;                         //   9,600 rows

  // 0) pack both weight matrices to bf16 fragment order (once per launch)
  pack_b_kernel<<<DD * DD / 256, 256, 0, stream>>>(w_val, packB1);
  pack_b_kernel<<<DD * DD / 256, 256, 0, stream>>>(w_out, packB2);

  // 1) value projection (dominant GEMM, bf16 WMMA, 32x64 tile per wave)
  wmma_gemm256_kernel<<<M1 / 32, 128, 0, stream>>>(value, packB1, b_val, v_ws, M1);
  // 2) offsets + attention softmax
  qproj_kernel<<<M2, 288, 0, stream>>>(query, w_off, b_off, w_attn, b_attn, off_ws, aw_ws);
  // 3) bilinear sampling + weighted sum
  msda_sample_kernel<<<M2, 256, 0, stream>>>(refp, off_ws, aw_ws, v_ws, att_ws);
  // 4) output projection (bf16 WMMA)
  wmma_gemm256_kernel<<<M2 / 32, 128, 0, stream>>>(att_ws, packB2, b_out, out, M2);
}
